// CrossViewSwapAttention_69269232550311
// MI455X (gfx1250) — matrix operand order repaired and checked by reading, not compiled, not run on hardware
//
#include <hip/hip_runtime.h>
#include <math.h>

#define DEV __device__ __forceinline__

typedef __attribute__((ext_vector_type(16))) __bf16        v16bf;
typedef __attribute__((ext_vector_type(8)))  float         v8f;
typedef __attribute__((ext_vector_type(8)))  unsigned int  v8u;
typedef unsigned short u16;
typedef unsigned int   u32;

// ---------------- problem constants ----------------
constexpr int B_ = 2, NV_ = 6, D_ = 128, H_ = 128, W_ = 128, FH_ = 64, FW_ = 64;
constexpr int L_ = 256;          // windows per batch (16x16)
constexpr int TQ_ = 384;         // query tokens per window  (n*8*8)
constexpr int TK_ = 96;          // key tokens per window    (n*4*4)
constexpr float LN_EPS_ = 1e-5f, BN_EPS_ = 1e-5f;

// ---------------- scalar helpers ----------------
DEV u16 f2bf(float f) {
  u32 u = __builtin_bit_cast(u32, f);
  u32 r = (u + 0x7FFFu + ((u >> 16) & 1u)) >> 16;   // round-to-nearest-even
  return (u16)r;
}
DEV float bf2f(u16 h) { u32 u = ((u32)h) << 16; return __builtin_bit_cast(float, u); }
DEV u32 pack2(u16 lo, u16 hi) { return (u32)lo | ((u32)hi << 16); }
DEV v8f vzero() {
  v8f z;
  #pragma unroll
  for (int i = 0; i < 8; ++i) z[i] = 0.f;
  return z;
}

DEV float rmax16(float v) {       // row reduce within half-wave (16 lanes)
  v = fmaxf(v, __shfl_xor(v, 1)); v = fmaxf(v, __shfl_xor(v, 2));
  v = fmaxf(v, __shfl_xor(v, 4)); v = fmaxf(v, __shfl_xor(v, 8)); return v;
}
DEV float rsum16(float v) {
  v += __shfl_xor(v, 1); v += __shfl_xor(v, 2);
  v += __shfl_xor(v, 4); v += __shfl_xor(v, 8); return v;
}

// ---------------- WMMA fragment helpers (bf16 16x16x32) ----------------
// A (16x32): lane m=l&15 holds row m; lanes 0-15: K={0..7,16..23}, 16-31: +8.
DEV v8u frag_a(const u16* src, int ld, int lane) {
  const u16* p = src + (lane & 15) * ld + ((lane >> 4) * 8);
  v8u r;
  #pragma unroll
  for (int j = 0; j < 8; ++j) {
    int k = (j < 4) ? (2 * j) : (2 * j + 8);
    r[j] = *(const u32*)(p + k);
  }
  return r;
}
// B (32x16) from n-major (transposed) storage Bt[n][k]; lanes 0-15: K={0..7,16..23},
// lanes 16-31: +8 (same order as A -> 8 dword loads).
DEV v8u frag_b(const u16* src, int ld, int lane) {
  const u16* p = src + (lane & 15) * ld + ((lane >> 4) * 8);
  v8u r;
  #pragma unroll
  for (int j = 0; j < 8; ++j) r[j] = *(const u32*)(p + ((j < 4) ? (2 * j) : (2 * j + 8)));
  return r;
}
DEV v8f wmma_bf16(v8u a, v8u b, v8f c) {
  v16bf av = __builtin_bit_cast(v16bf, a);
  v16bf bv = __builtin_bit_cast(v16bf, b);
  return __builtin_amdgcn_wmma_f32_16x16x32_bf16(false, av, false, bv,
                                                 (short)0, c, false, false);
}

// ================= kernel 1: camera-center embedding (12 x 128) =================
__global__ void k_cembed(const float* __restrict__ Wcam, const float* __restrict__ Einv,
                         float* __restrict__ cemb) {
  int idx = blockIdx.x * 256 + threadIdx.x;
  if (idx >= B_ * NV_ * D_) return;
  int bn = idx >> 7, d = idx & 127;
  const float* E = Einv + bn * 16;
  float s = 0.f;
  #pragma unroll
  for (int k = 0; k < 4; ++k) s += Wcam[d * 4 + k] * E[k * 4 + 3];
  cemb[idx] = s;
}

// ================= kernel 2: BEV query base  q = norm(w_embed - c_embed) + x ====
__global__ void k_build_q(const float* __restrict__ x, const float* __restrict__ Wbev,
                          const float* __restrict__ bbev, const float* __restrict__ cemb,
                          u16* __restrict__ qbase) {
  int wid = blockIdx.x * 8 + (threadIdx.x >> 5);   // one wave per (bn,h,w) pixel
  int lane = threadIdx.x & 31;
  int w = wid & 127, h = (wid >> 7) & 127, bn = wid >> 14;
  int b = bn / NV_;
  float g0 = (128.f - 256.f * (float)h / 127.f) * (1.f / 2.56f);
  float g1 = (128.f - 256.f * (float)w / 127.f) * (1.f / 2.56f);
  int d0 = lane * 4;
  float e[4]; float ss = 0.f;
  #pragma unroll
  for (int t = 0; t < 4; ++t) {
    int d = d0 + t;
    float v = Wbev[d * 2 + 0] * g0 + Wbev[d * 2 + 1] * g1 + bbev[d] - cemb[bn * 128 + d];
    e[t] = v; ss += v * v;
  }
  for (int m = 1; m < 32; m <<= 1) ss += __shfl_xor(ss, m);
  float rn = 1.f / (sqrtf(ss) + 1e-7f);
  float xv[4];
  #pragma unroll
  for (int t = 0; t < 4; ++t)
    xv[t] = x[(((size_t)b * D_ + d0 + t) * H_ + h) * W_ + w];
  u16* o = qbase + (size_t)wid * 128 + d0;
  *(u32*)(o)     = pack2(f2bf(e[0] * rn + xv[0]), f2bf(e[1] * rn + xv[1]));
  *(u32*)(o + 2) = pack2(f2bf(e[2] * rn + xv[2]), f2bf(e[3] * rn + xv[3]));
}

// ================= kernel 3: image ray embed + BN-ReLU activations ==============
__global__ void k_build_cam(const float* __restrict__ feat, const float* __restrict__ Iinv,
                            const float* __restrict__ Einv, const float* __restrict__ Wimg,
                            const float* __restrict__ cemb,
                            const float* __restrict__ g1, const float* __restrict__ b1,
                            const float* __restrict__ g2, const float* __restrict__ b2,
                            u16* __restrict__ imge, u16* __restrict__ actp,
                            u16* __restrict__ actl) {
  int wid = blockIdx.x * 8 + (threadIdx.x >> 5);   // wave per (bn, pixel)
  int lane = threadIdx.x & 31;
  int p = wid & 4095, bn = wid >> 12;
  int hh = p >> 6, ww = p & 63;
  float px = (float)ww * (512.f / 63.f), py = (float)hh * (512.f / 63.f);
  const float* I = Iinv + bn * 9; const float* E = Einv + bn * 16;
  float c0 = I[0] * px + I[1] * py + I[2];
  float c1 = I[3] * px + I[4] * py + I[5];
  float c2 = I[6] * px + I[7] * py + I[8];
  float dfl[4];
  #pragma unroll
  for (int i = 0; i < 4; ++i)
    dfl[i] = E[i * 4 + 0] * c0 + E[i * 4 + 1] * c1 + E[i * 4 + 2] * c2 + E[i * 4 + 3];
  int d0 = lane * 4;
  float e[4]; float ss = 0.f;
  #pragma unroll
  for (int t = 0; t < 4; ++t) {
    int d = d0 + t;
    float v = Wimg[d * 4 + 0] * dfl[0] + Wimg[d * 4 + 1] * dfl[1] +
              Wimg[d * 4 + 2] * dfl[2] + Wimg[d * 4 + 3] * dfl[3] - cemb[bn * 128 + d];
    e[t] = v; ss += v * v;
  }
  for (int m = 1; m < 32; m <<= 1) ss += __shfl_xor(ss, m);
  float rn = 1.f / (sqrtf(ss) + 1e-7f);
  size_t ro = (size_t)wid * 128 + d0;
  *(u32*)(imge + ro)     = pack2(f2bf(e[0] * rn), f2bf(e[1] * rn));
  *(u32*)(imge + ro + 2) = pack2(f2bf(e[2] * rn), f2bf(e[3] * rn));
  float rb = rsqrtf(1.f + BN_EPS_);
  float a[4], l[4];
  #pragma unroll
  for (int t = 0; t < 4; ++t) {
    int c = d0 + t;
    float f = feat[((size_t)bn * 128 + c) * 4096 + p];
    a[t] = fmaxf(f * (g1[c] * rb) + b1[c], 0.f);
    l[t] = fmaxf(f * (g2[c] * rb) + b2[c], 0.f);
  }
  *(u32*)(actp + ro)     = pack2(f2bf(a[0]), f2bf(a[1]));
  *(u32*)(actp + ro + 2) = pack2(f2bf(a[2]), f2bf(a[3]));
  *(u32*)(actl + ro)     = pack2(f2bf(l[0]), f2bf(l[1]));
  *(u32*)(actl + ro + 2) = pack2(f2bf(l[2]), f2bf(l[3]));
}

// ================= kernel 4: generic rows x (K=128) x (N=128) WMMA GEMM =========
// out[r,d] = sum_c A[r,c] * Wdc[d,c]  (+addsrc)   (Wdc already n-major/k-contig)
__global__ void k_gemm128(const u16* __restrict__ A, const float* __restrict__ Wdc,
                          const u16* __restrict__ addsrc, u16* __restrict__ out) {
  __shared__ u16 Wt_s[128 * 128];
  int tid = threadIdx.x, lane = tid & 31, wave = tid >> 5;
  for (int i = tid; i < 16384; i += 128) Wt_s[i] = f2bf(Wdc[i]);
  __syncthreads();
  size_t row0 = (size_t)blockIdx.x * 64 + wave * 16;
  v8f acc[8];
  #pragma unroll
  for (int i = 0; i < 8; ++i) acc[i] = vzero();
  #pragma unroll
  for (int kc = 0; kc < 4; ++kc) {
    v8u a = frag_a(A + row0 * 128 + kc * 32, 128, lane);
    #pragma unroll
    for (int nt = 0; nt < 8; ++nt) {
      v8u bf = frag_b(Wt_s + nt * 16 * 128 + kc * 32, 128, lane);
      acc[nt] = wmma_bf16(a, bf, acc[nt]);
    }
  }
  int half = lane >> 4, cn = lane & 15;
  #pragma unroll
  for (int nt = 0; nt < 8; ++nt) {
    int col = nt * 16 + cn;
    #pragma unroll
    for (int j = 0; j < 8; ++j) {
      size_t r = row0 + j + 8 * half;
      float v = acc[nt][j];
      if (addsrc) v += bf2f(addsrc[r * 128 + col]);
      out[r * 128 + col] = f2bf(v);
    }
  }
}

// ================= kernel 5: token-gather + LayerNorm + projection ==============
// dst[row,:] = LN(src[map(row),:]) @ Wkn + bias     (Wkn is [128 in][128 out])
__global__ void k_ln_proj(const u16* __restrict__ src, const float* __restrict__ lnG,
                          const float* __restrict__ lnB, const float* __restrict__ Wkn,
                          const float* __restrict__ bias, u16* __restrict__ dst,
                          int TPL, int SH, int SW, int w1, int w2, int mode, int bcastN) {
  __shared__ u16 Wt_s[128 * 128];
  __shared__ u16 tile_s[4][16 * 128];
  __shared__ float g_s[128], b_s[128], bias_s[128];
  int tid = threadIdx.x, lane = tid & 31, wave = tid >> 5;
  for (int i = tid; i < 16384; i += 128) {
    int k = i & 127, n = i >> 7;
    Wt_s[n * 128 + k] = f2bf(Wkn[k * 128 + n]);       // transpose-stage
  }
  if (tid < 128) { g_s[tid] = lnG[tid]; b_s[tid] = lnB[tid]; bias_s[tid] = bias[tid]; }
  __syncthreads();
  int half = lane >> 4, r = lane & 15;
  int GY = SW / w2, GX1 = SH / w1, GX2 = SW / w2, ww12 = w1 * w2;
  size_t row = (size_t)blockIdx.x * 64 + wave * 16 + r;
  int LT = L_ * TPL;
  int b = (int)(row / LT); int rem = (int)(row % LT);
  int l = rem / TPL, t = rem % TPL;
  int nn = t / ww12, rr = t % ww12, i2 = rr / w2, j2 = rr % w2;
  int X = l / GY, Y = l % GY;
  int h, w;
  if (mode == 0) { h = X * w1 + i2; w = Y * w2 + j2; }
  else           { h = i2 * GX1 + X; w = j2 * GX2 + Y; }
  size_t srow = bcastN ? (((size_t)b * SH + h) * SW + w)
                       : ((((size_t)b * NV_ + nn) * SH + h) * SW + w);
  const u16* sp = src + srow * 128 + half * 64;
  float vals[64]; float s = 0.f, sq = 0.f;
  #pragma unroll
  for (int k = 0; k < 64; ++k) { float v = bf2f(sp[k]); vals[k] = v; s += v; sq += v * v; }
  s  += __shfl_xor(s, 16);
  sq += __shfl_xor(sq, 16);
  float mu = s * (1.f / 128.f);
  float var = sq * (1.f / 128.f) - mu * mu;
  float rs = rsqrtf(var + LN_EPS_);
  u16* tp = &tile_s[wave][r * 128 + half * 64];
  #pragma unroll
  for (int k = 0; k < 64; ++k) {
    int e = half * 64 + k;
    tp[k] = f2bf((vals[k] - mu) * rs * g_s[e] + b_s[e]);
  }
  asm volatile("s_wait_dscnt 0" ::: "memory");        // wave-local LDS fence
  v8f acc[8];
  #pragma unroll
  for (int i = 0; i < 8; ++i) acc[i] = vzero();
  #pragma unroll
  for (int kc = 0; kc < 4; ++kc) {
    v8u a = frag_a(&tile_s[wave][kc * 32], 128, lane);
    #pragma unroll
    for (int nt = 0; nt < 8; ++nt) {
      v8u bf = frag_b(Wt_s + nt * 16 * 128 + kc * 32, 128, lane);
      acc[nt] = wmma_bf16(a, bf, acc[nt]);
    }
  }
  size_t row0 = (size_t)blockIdx.x * 64 + wave * 16;
  #pragma unroll
  for (int nt = 0; nt < 8; ++nt) {
    int col = nt * 16 + (lane & 15);
    #pragma unroll
    for (int j = 0; j < 8; ++j)
      dst[(row0 + j + 8 * half) * 128 + col] = f2bf(acc[nt][j] + bias_s[col]);
  }
}

// ================= kernel 6: per-window multi-head attention ====================
__global__ void k_attn(const u16* __restrict__ qh, const u16* __restrict__ kh,
                       const u16* __restrict__ vh, u16* __restrict__ aout) {
  __shared__ u16 kh_s[TK_ * 128];      // K row-major  (24.5 KB)
  __shared__ u16 vt_s[128 * TK_];      // V transposed (24.5 KB)
  __shared__ u16 sc_s[4][16 * TK_];    // per-wave score tiles (12 KB)
  int tid = threadIdx.x, lane = tid & 31, wave = tid >> 5;
  size_t bl = blockIdx.x;
  size_t kb = bl * TK_ * 128;
  const u32* ksrc = (const u32*)(kh + kb);
  u32* kdst = (u32*)kh_s;
  for (int i = tid; i < TK_ * 64; i += 128) kdst[i] = ksrc[i];
  for (int i = tid; i < TK_ * 128; i += 128) {
    int t = i >> 7, d = i & 127;
    vt_s[d * TK_ + t] = vh[kb + i];
  }
  __syncthreads();
  size_t qb = bl * TQ_ * 128;
  const float scale = 0.17677669529663687f;   // 1/sqrt(32)
  int half = lane >> 4, cn = lane & 15;
  for (int qt = wave; qt < 24; qt += 4) {     // 24 q-tiles of 16 tokens
    size_t qrow = qb + (size_t)qt * 16 * 128;
    #pragma unroll
    for (int hd = 0; hd < 4; ++hd) {
      v8u aq = frag_a(qh + qrow + hd * 32, 128, lane);
      v8f s[6];
      #pragma unroll
      for (int nt = 0; nt < 6; ++nt) {
        v8u bf = frag_b(kh_s + nt * 16 * 128 + hd * 32, 128, lane);
        s[nt] = wmma_bf16(aq, bf, vzero());
      }
      float mx[8], sm[8];
      #pragma unroll
      for (int j = 0; j < 8; ++j) {
        float m = -1e30f;
        #pragma unroll
        for (int nt = 0; nt < 6; ++nt) m = fmaxf(m, s[nt][j] * scale);
        mx[j] = rmax16(m);
      }
      #pragma unroll
      for (int j = 0; j < 8; ++j) {
        float acc = 0.f;
        #pragma unroll
        for (int nt = 0; nt < 6; ++nt) {
          float e = __expf(s[nt][j] * scale - mx[j]);
          s[nt][j] = e; acc += e;
        }
        sm[j] = 1.f / rsum16(acc);
      }
      #pragma unroll
      for (int nt = 0; nt < 6; ++nt)
        #pragma unroll
        for (int j = 0; j < 8; ++j)
          sc_s[wave][(j + 8 * half) * TK_ + nt * 16 + cn] = f2bf(s[nt][j] * sm[j]);
      asm volatile("s_wait_dscnt 0" ::: "memory");
      v8f o0 = vzero(), o1 = vzero();
      #pragma unroll
      for (int kc = 0; kc < 3; ++kc) {
        v8u af  = frag_a(&sc_s[wave][kc * 32], TK_, lane);
        v8u bf0 = frag_b(vt_s + (hd * 32) * TK_ + kc * 32, TK_, lane);
        v8u bf1 = frag_b(vt_s + (hd * 32 + 16) * TK_ + kc * 32, TK_, lane);
        o0 = wmma_bf16(af, bf0, o0);
        o1 = wmma_bf16(af, bf1, o1);
      }
      #pragma unroll
      for (int j = 0; j < 8; ++j) {
        size_t ro = qb + (size_t)(qt * 16 + j + 8 * half) * 128;
        aout[ro + hd * 32 + cn]      = f2bf(o0[j]);
        aout[ro + hd * 32 + 16 + cn] = f2bf(o1[j]);
      }
    }
  }
}

// ================= kernel 7: output projection + camera mean + skip =============
__global__ void k_proj_mean_skip(const u16* __restrict__ aout, const float* __restrict__ Wp,
                                 const float* __restrict__ bp, const float* __restrict__ skip,
                                 int skipChFirst, float* __restrict__ outbuf, int mode) {
  __shared__ u16 Wt_s[128 * 128];
  __shared__ float bp_s[128];
  int tid = threadIdx.x, lane = tid & 31, wave = tid >> 5;
  for (int i = tid; i < 16384; i += 128) {
    int k = i & 127, n = i >> 7;
    Wt_s[n * 128 + k] = f2bf(Wp[k * 128 + n]);
  }
  if (tid < 128) bp_s[tid] = bp[tid];
  __syncthreads();
  int bl = blockIdx.x, b = bl >> 8, l = bl & 255;
  int X = l >> 4, Y = l & 15;
  size_t ab = (size_t)bl * TQ_ * 128;
  int pos0 = wave * 16;
  v8f acc[8];
  #pragma unroll
  for (int i = 0; i < 8; ++i) acc[i] = vzero();
  for (int nn = 0; nn < 6; ++nn) {                 // sum over cameras
    #pragma unroll
    for (int kc = 0; kc < 4; ++kc) {
      v8u a = frag_a(aout + ab + (size_t)(nn * 64 + pos0) * 128 + kc * 32, 128, lane);
      #pragma unroll
      for (int nt = 0; nt < 8; ++nt) {
        v8u bf = frag_b(Wt_s + nt * 16 * 128 + kc * 32, 128, lane);
        acc[nt] = wmma_bf16(a, bf, acc[nt]);
      }
    }
  }
  int half = lane >> 4, cn = lane & 15;
  #pragma unroll
  for (int nt = 0; nt < 8; ++nt) {
    int col = nt * 16 + cn;
    #pragma unroll
    for (int j = 0; j < 8; ++j) {
      int pos = pos0 + j + 8 * half;
      int i2 = pos >> 3, j2 = pos & 7;
      int h, w;
      if (mode == 0) { h = X * 8 + i2; w = Y * 8 + j2; }
      else           { h = i2 * 16 + X; w = j2 * 16 + Y; }
      float sk = skipChFirst ? skip[(((size_t)b * 128 + col) * H_ + h) * W_ + w]
                             : skip[(((size_t)b * H_ + h) * W_ + w) * 128 + col];
      outbuf[(((size_t)b * H_ + h) * W_ + w) * 128 + col] =
          acc[nt][j] * (1.f / 6.f) + bp_s[col] + sk;
    }
  }
}

// ================= kernel 8: x += MLP(LN(x))  (128 -> 512 -> 128) ===============
__global__ void k_mlp(float* __restrict__ buf, const float* __restrict__ lnG,
                      const float* __restrict__ lnB, const float* __restrict__ W1,
                      const float* __restrict__ b1, const float* __restrict__ W2,
                      const float* __restrict__ b2) {
  __shared__ u16 xln_s[32 * 128];
  __shared__ u16 h_s[32 * 128];
  __shared__ u16 Wt_s[128 * 128];
  int tid = threadIdx.x, lane = tid & 31, wave = tid >> 5;
  size_t rbase = (size_t)blockIdx.x * 32;
  if (tid < 32) {
    const float* rp = buf + (rbase + tid) * 128;
    float s = 0.f, sq = 0.f;
    for (int e = 0; e < 128; ++e) { float v = rp[e]; s += v; sq += v * v; }
    float mu = s / 128.f, var = sq / 128.f - mu * mu, rs = rsqrtf(var + LN_EPS_);
    for (int e = 0; e < 128; ++e)
      xln_s[tid * 128 + e] = f2bf((rp[e] - mu) * rs * lnG[e] + lnB[e]);
  }
  int rt = wave >> 1, nh = wave & 1;
  int half = lane >> 4, cn = lane & 15;
  v8f oa[4];
  #pragma unroll
  for (int i = 0; i < 4; ++i) oa[i] = vzero();
  for (int nc = 0; nc < 4; ++nc) {                   // hidden chunks of 128
    __syncthreads();
    for (int i = tid; i < 16384; i += 128) {
      int k = i & 127, n = i >> 7;
      Wt_s[n * 128 + k] = f2bf(W1[k * 512 + nc * 128 + n]);
    }
    __syncthreads();
    v8f ha[4];
    #pragma unroll
    for (int i = 0; i < 4; ++i) ha[i] = vzero();
    #pragma unroll
    for (int kc = 0; kc < 4; ++kc) {
      v8u a = frag_a(xln_s + rt * 16 * 128 + kc * 32, 128, lane);
      #pragma unroll
      for (int nt = 0; nt < 4; ++nt) {
        v8u bf = frag_b(Wt_s + (nh * 64 + nt * 16) * 128 + kc * 32, 128, lane);
        ha[nt] = wmma_bf16(a, bf, ha[nt]);
      }
    }
    #pragma unroll
    for (int nt = 0; nt < 4; ++nt) {
      int cl = nh * 64 + nt * 16 + cn;
      #pragma unroll
      for (int j = 0; j < 8; ++j) {
        float v = ha[nt][j] + b1[nc * 128 + cl];
        float g = 0.5f * v * (1.f + erff(v * 0.70710678f));   // exact GELU
        h_s[(rt * 16 + j + 8 * half) * 128 + cl] = f2bf(g);
      }
    }
    __syncthreads();
    for (int i = tid; i < 16384; i += 128) {
      int k = i & 127, n = i >> 7;
      Wt_s[n * 128 + k] = f2bf(W2[(nc * 128 + k) * 128 + n]);
    }
    __syncthreads();
    #pragma unroll
    for (int kc = 0; kc < 4; ++kc) {
      v8u a = frag_a(h_s + rt * 16 * 128 + kc * 32, 128, lane);
      #pragma unroll
      for (int nt = 0; nt < 4; ++nt) {
        v8u bf = frag_b(Wt_s + (nh * 64 + nt * 16) * 128 + kc * 32, 128, lane);
        oa[nt] = wmma_bf16(a, bf, oa[nt]);
      }
    }
  }
  #pragma unroll
  for (int nt = 0; nt < 4; ++nt) {
    int cl = nh * 64 + nt * 16 + cn;
    #pragma unroll
    for (int j = 0; j < 8; ++j) {
      size_t r = rbase + rt * 16 + j + 8 * half;
      buf[r * 128 + cl] += oa[nt][j] + b2[cl];
    }
  }
}

// ================= kernel 9a: f32 -> bf16 conversion ============================
__global__ void k_cvt_bf16(const float* __restrict__ src, u16* __restrict__ dst, int n) {
  int i = blockIdx.x * 256 + threadIdx.x;
  int idx = i * 2;
  if (idx + 1 < n)
    *(u32*)(dst + idx) = pack2(f2bf(src[idx]), f2bf(src[idx + 1]));
}

// ================= kernel 9b: final LayerNorm + NHWC->NCHW ======================
__global__ void k_post_ln(const float* __restrict__ q3, const float* __restrict__ g,
                          const float* __restrict__ bta, float* __restrict__ out) {
  int wid = blockIdx.x * 8 + (threadIdx.x >> 5);
  int lane = threadIdx.x & 31;
  const float* rp = q3 + (size_t)wid * 128;
  int e0 = lane * 4;
  float v[4]; float s = 0.f, sq = 0.f;
  #pragma unroll
  for (int t = 0; t < 4; ++t) { v[t] = rp[e0 + t]; s += v[t]; sq += v[t] * v[t]; }
  for (int m = 1; m < 32; m <<= 1) { s += __shfl_xor(s, m); sq += __shfl_xor(sq, m); }
  float mu = s / 128.f, var = sq / 128.f - mu * mu, rs = rsqrtf(var + LN_EPS_);
  int b = wid >> 14, h = (wid >> 7) & 127, w = wid & 127;
  #pragma unroll
  for (int t = 0; t < 4; ++t) {
    int d = e0 + t;
    out[(((size_t)b * 128 + d) * 128 + h) * 128 + w] = (v[t] - mu) * rs * g[d] + bta[d];
  }
}

// ============================ launcher ==========================================
extern "C" void kernel_launch(void* const* d_in, const int* in_sizes, int n_in,
                              void* d_out, int out_size, void* d_ws, size_t ws_size,
                              hipStream_t stream) {
  (void)in_sizes; (void)n_in; (void)out_size; (void)ws_size;
  // Input indices assume recursive insertion-order flattening of setup_inputs().
  auto P = [&](int i) { return (const float*)d_in[i]; };
  const float* x    = P(0);
  const float* feat = P(1);
  const float* Iinv = P(2);
  const float* Einv = P(3);
  const float* Wcam = P(4);  const float* Wimg = P(5);
  const float* Wbev = P(6);  const float* bbev = P(7);
  const float* fp_g = P(8);  const float* fp_b = P(9);  const float* fp_W = P(10);
  const float* fl_g = P(11); const float* fl_b = P(12); const float* fl_W = P(13);
  // attn1: 14..27 ; attn2: 28..41 ; pn1: 42,43 ; mlp1: 44..47 ; pn2: 48,49 ;
  // mlp2: 50..53 ; post: 54,55 ; index: 56 (ignored)

  char* ws = (char*)d_ws;
  size_t off = 0;
  auto take = [&](size_t bytes) -> char* {
    char* p = ws + off; off += (bytes + 255) & ~(size_t)255; return p;
  };
  float* cemb = (float*)take((size_t)12 * 128 * 4);
  u16* qbase  = (u16*)take((size_t)B_ * NV_ * H_ * W_ * 128 * 2);
  u16* kbase  = (u16*)take((size_t)B_ * NV_ * FH_ * FW_ * 128 * 2);
  u16* vbase  = (u16*)take((size_t)B_ * NV_ * FH_ * FW_ * 128 * 2);
  u16* qhB    = (u16*)take((size_t)B_ * L_ * TQ_ * 128 * 2);
  u16* khB    = (u16*)take((size_t)B_ * L_ * TK_ * 128 * 2);
  u16* vhB    = (u16*)take((size_t)B_ * L_ * TK_ * 128 * 2);
  u16* aoutB  = (u16*)take((size_t)B_ * L_ * TQ_ * 128 * 2);
  float* q1   = (float*)take((size_t)B_ * H_ * W_ * 128 * 4);
  float* q3   = (float*)take((size_t)B_ * H_ * W_ * 128 * 4);
  // aliases into regions that are dead by the time the alias is written:
  u16* imge = aoutB;                                   // dead once kbase built
  u16* actp = aoutB + (size_t)12 * 4096 * 128;
  u16* actl = actp  + (size_t)12 * 4096 * 128;
  u16* q1bf = qbase;                                   // qbase dead after stage-1 qh

  // ---- embeddings + features ----
  k_cembed<<<6, 256, 0, stream>>>(Wcam, Einv, cemb);
  k_build_q<<<(B_ * NV_ * H_ * W_) / 8, 256, 0, stream>>>(x, Wbev, bbev, cemb, qbase);
  k_build_cam<<<(B_ * NV_ * FH_ * FW_) / 8, 256, 0, stream>>>(
      feat, Iinv, Einv, Wimg, cemb, fp_g, fp_b, fl_g, fl_b, imge, actp, actl);
  k_gemm128<<<768, 128, 0, stream>>>(actp, fp_W, imge, kbase);        // key
  k_gemm128<<<768, 128, 0, stream>>>(actl, fl_W, nullptr, vbase);     // val

  // ---- stage 1: local-window cross-attention ----
  k_ln_proj<<<3072, 128, 0, stream>>>(qbase, P(14), P(15), P(16), P(17), qhB,
                                      TQ_, H_, W_, 8, 8, /*mode*/0, /*bcast*/0);
  k_ln_proj<<<768, 128, 0, stream>>>(kbase, P(18), P(19), P(20), P(21), khB,
                                     TK_, FH_, FW_, 4, 4, 0, 0);
  k_ln_proj<<<768, 128, 0, stream>>>(vbase, P(22), P(23), P(24), P(25), vhB,
                                     TK_, FH_, FW_, 4, 4, 0, 0);
  k_attn<<<B_ * L_, 128, 0, stream>>>(qhB, khB, vhB, aoutB);
  k_proj_mean_skip<<<B_ * L_, 128, 0, stream>>>(aoutB, P(26), P(27), x, /*chFirst*/1,
                                                q1, /*mode*/0);
  k_mlp<<<(B_ * H_ * W_) / 32, 128, 0, stream>>>(q1, P(42), P(43), P(44), P(45),
                                                 P(46), P(47));

  // ---- stage 2: grid (dilated) cross-attention ----
  k_cvt_bf16<<<(B_ * H_ * W_ * 128) / 512, 256, 0, stream>>>(q1, q1bf,
                                                             B_ * H_ * W_ * 128);
  k_ln_proj<<<3072, 128, 0, stream>>>(q1bf, P(28), P(29), P(30), P(31), qhB,
                                      TQ_, H_, W_, 8, 8, /*mode*/1, /*bcast*/1);
  k_ln_proj<<<768, 128, 0, stream>>>(kbase, P(32), P(33), P(34), P(35), khB,
                                     TK_, FH_, FW_, 4, 4, 1, 0);
  k_ln_proj<<<768, 128, 0, stream>>>(vbase, P(36), P(37), P(38), P(39), vhB,
                                     TK_, FH_, FW_, 4, 4, 1, 0);
  k_attn<<<B_ * L_, 128, 0, stream>>>(qhB, khB, vhB, aoutB);
  k_proj_mean_skip<<<B_ * L_, 128, 0, stream>>>(aoutB, P(40), P(41), q1, /*chFirst*/0,
                                                q3, /*mode*/1);
  k_mlp<<<(B_ * H_ * W_) / 32, 128, 0, stream>>>(q3, P(48), P(49), P(50), P(51),
                                                 P(52), P(53));

  // ---- final LN + NCHW transpose ----
  k_post_ln<<<(B_ * H_ * W_) / 8, 256, 0, stream>>>(q3, P(54), P(55), (float*)d_out);
}
